// MultiheadAttentionMechanism_30537217475344
// MI455X (gfx1250) — matrix operand order repaired and checked by reading, not compiled, not run on hardware
//
#include <hip/hip_runtime.h>
#include <hip/hip_bf16.h>

typedef __attribute__((ext_vector_type(16))) _Float16 v16h;
typedef __attribute__((ext_vector_type(8)))  _Float16 v8h;
typedef __attribute__((ext_vector_type(4)))  _Float16 v4h;
typedef __attribute__((ext_vector_type(8)))  float    v8f;
typedef __attribute__((ext_vector_type(4)))  float    v4f;

#define WMMA_F16(a,b,c) __builtin_amdgcn_wmma_f32_16x16x32_f16(false,(a),false,(b),(short)0,(c),false,false)

static constexpr int BATCH  = 8;
static constexpr int LEN    = 1024;
static constexpr int DMODEL = 512;
static constexpr int NHEAD  = 8;
static constexpr int DK     = 64;
static constexpr int PSTR   = 1028;   // LDS row stride (floats): 16B-aligned, kills bank conflicts

static __device__ __forceinline__ v16h cat16(v8h lo, v8h hi) {
  return __builtin_shufflevector(lo, hi, 0,1,2,3,4,5,6,7,8,9,10,11,12,13,14,15);
}

// ---------------------------------------------------------------------------
// Kernel 1: convert + transpose four 512x512 f32 weights -> f16 WT[n][k]
// (B-fragments become contiguous 16B loads). Layout [4][512][512]: q,k,v,o.
// ---------------------------------------------------------------------------
__global__ void k_wtrans(const float* __restrict__ Wq, const float* __restrict__ Wk,
                         const float* __restrict__ Wv, const float* __restrict__ Wo,
                         _Float16* __restrict__ WT) {
  int idx = blockIdx.x * blockDim.x + threadIdx.x;   // 4*512*512 threads
  int w = idx >> 18;
  int r = idx & 262143;
  int n = r >> 9, k = r & 511;
  const float* W = (w == 0) ? Wq : (w == 1) ? Wk : (w == 2) ? Wv : Wo;
  WT[idx] = (_Float16)W[k * 512 + n];
}

// ---------------------------------------------------------------------------
// Kernel 2: one-shot f32 -> f16 conversion of query/key/value activations so
// the projection GEMMs re-read half the bytes and skip per-iteration cvts.
// out layout: [3][8192*512], order q,k,v. 4 elements per thread.
// ---------------------------------------------------------------------------
__global__ void k_xcvt(const float* __restrict__ q, const float* __restrict__ k,
                       const float* __restrict__ v, _Float16* __restrict__ out) {
  int t = blockIdx.x * blockDim.x + threadIdx.x;     // 3*1048576 threads
  int w = t >> 20;
  int r = t & 1048575;
  const float* X = (w == 0) ? q : (w == 1) ? k : v;
  v4f x = *(const v4f*)(X + (size_t)r * 4);
  v4h y;
#pragma unroll
  for (int e = 0; e < 4; ++e) y[e] = (_Float16)x[e];
  *(v4h*)(out + (size_t)t * 4) = y;
}

// ---------------------------------------------------------------------------
// Kernel 3: unified WMMA GEMM  Y = A[8192x512](f16) @ W[512x512]
// One wave per 16x16 tile, K-loop step 32 (v_wmma_f32_16x16x32_f16).
//   mode 0: f16 out, head-major  Y[b][h][l][dk]   (Q, K)
//   mode 1: f16 out, transposed  Y[b][h][dk][l]   (V)
//   mode 2: f32 out + bias, flat [row][col]       (final projection)
// ---------------------------------------------------------------------------
__global__ void k_gemm(const _Float16* __restrict__ A, const _Float16* __restrict__ WT,
                       _Float16* __restrict__ Yh, float* __restrict__ Yf,
                       const float* __restrict__ bo, int mode) {
  int wave = (blockIdx.x * blockDim.x + threadIdx.x) >> 5;
  int lane = threadIdx.x & 31;
  int tr = wave >> 5, tc = wave & 31;
  int ln = lane & 15, h2 = lane >> 4;

  const _Float16* arow = A  + (size_t)(tr * 16 + ln) * 512;
  const _Float16* wrow = WT + (size_t)(tc * 16 + ln) * 512;

  v8f c = {};
  for (int k0 = 0; k0 < 512; k0 += 32) {
    v16h a  = cat16(*(const v8h*)(arow + k0 + h2 * 8), *(const v8h*)(arow + k0 + 16 + h2 * 8));
    v16h bf = cat16(*(const v8h*)(wrow + k0 + h2 * 8), *(const v8h*)(wrow + k0 + 16 + h2 * 8));
    c = WMMA_F16(a, bf, c);
  }

#pragma unroll
  for (int r = 0; r < 8; ++r) {
    int mm  = r + 8 * h2;
    int row = tr * 16 + mm;            // = b*1024 + l
    int col = tc * 16 + ln;            // = h*64 + dk
    if (mode == 2) {
      Yf[(size_t)row * 512 + col] = c[r] + bo[col];
    } else {
      int bb = row >> 10, ll = row & 1023;
      int hh = col >> 6,  dk = col & 63;
      size_t o = (mode == 0)
               ? (((size_t)(bb * 8 + hh) * 1024 + ll) * 64 + dk)
               : (((size_t)(bb * 8 + hh) * 64 + dk) * 1024 + ll);
      Yh[o] = (_Float16)c[r];
    }
  }
}

// ---------------------------------------------------------------------------
// Kernel 4: attention core. 4 waves/block, each wave owns a 16-row q strip:
//   scores (128 WMMA) -> padded LDS strip -> wave-parallel exact softmax
//   (normalized in place) -> aw streamed out via global_store_async_from_lds
//   -> P@V (128 WMMA) -> cv f16 to workspace.
// Dynamic LDS: 4 * 16 * 1028 f32 = 263168 B (< 320KB WGP limit).
// ---------------------------------------------------------------------------
__global__ void k_attn(const _Float16* __restrict__ Qh, const _Float16* __restrict__ Kh,
                       const _Float16* __restrict__ Vt, const int* __restrict__ klens,
                       const int* __restrict__ qlens, float* __restrict__ aw_out,
                       _Float16* __restrict__ cvws) {
  extern __shared__ float lds[];
  int wid  = threadIdx.x >> 5, lane = threadIdx.x & 31;
  float* P = lds + wid * (16 * PSTR);

  int gid   = blockIdx.x * 4 + wid;    // 0..4095 strips
  int bh    = gid >> 6;                // b*8 + h
  int b     = bh >> 3;
  int qb    = (gid & 63) * 16;
  int klen  = klens[b], qlen = qlens[b];
  int ln = lane & 15, h2 = lane >> 4;

  // --- Q A-fragments (rows fixed for the strip), DK=64 -> two frags ---
  const _Float16* qrow = Qh + (((size_t)bh * 1024) + qb + ln) * 64;
  v16h a0 = cat16(*(const v8h*)(qrow + h2 * 8),      *(const v8h*)(qrow + 16 + h2 * 8));
  v16h a1 = cat16(*(const v8h*)(qrow + 32 + h2 * 8), *(const v8h*)(qrow + 48 + h2 * 8));

  // --- scores e = (Q K^T) * dk^-0.5, masked, into LDS ---
  const _Float16* kbh = Kh + (size_t)bh * 1024 * 64;
  for (int kt = 0; kt < 64; ++kt) {
    const _Float16* krow = kbh + (size_t)(kt * 16 + ln) * 64;
    if (kt < 63) __builtin_prefetch(krow + 16 * 64, 0, 0);  // next K tile -> global_prefetch
    v16h b0 = cat16(*(const v8h*)(krow + h2 * 8),      *(const v8h*)(krow + 16 + h2 * 8));
    v16h b1 = cat16(*(const v8h*)(krow + 32 + h2 * 8), *(const v8h*)(krow + 48 + h2 * 8));
    v8f c = {};
    c = WMMA_F16(a0, b0, c);
    c = WMMA_F16(a1, b1, c);
#pragma unroll
    for (int r = 0; r < 8; ++r) {
      int mm  = r + 8 * h2;
      int col = kt * 16 + ln;
      bool valid = (col < klen) && ((qb + mm) < qlen);
      P[mm * PSTR + col] = valid ? c[r] * 0.125f : -1e9f;
    }
  }

  // --- exact softmax, all 32 lanes per row, normalized in place ---
  for (int r = 0; r < 16; ++r) {
    float* row = P + r * PSTR;
    float mx = -3.0e38f;
    for (int cc = lane * 4; cc < 1024; cc += 128) {
      v4f x = *(const v4f*)(row + cc);
      mx = fmaxf(mx, fmaxf(fmaxf(x[0], x[1]), fmaxf(x[2], x[3])));
    }
#pragma unroll
    for (int off = 16; off > 0; off >>= 1) mx = fmaxf(mx, __shfl_xor(mx, off, 32));
    float s = 0.0f;
    for (int cc = lane * 4; cc < 1024; cc += 128) {
      v4f x = *(const v4f*)(row + cc);
      v4f e;
#pragma unroll
      for (int j = 0; j < 4; ++j) { e[j] = __expf(x[j] - mx); s += e[j]; }
      *(v4f*)(row + cc) = e;
    }
#pragma unroll
    for (int off = 16; off > 0; off >>= 1) s += __shfl_xor(s, off, 32);
    float rinv = 1.0f / s;
    for (int cc = lane * 4; cc < 1024; cc += 128) {
      v4f x = *(const v4f*)(row + cc);
      x *= rinv;
      *(v4f*)(row + cc) = x;
    }
  }

  // --- stream normalized aw (the 256MB bandwidth limiter) straight from LDS
  //     via the CDNA5 async store path: no VGPR round-trip. ---
  {
    asm volatile("s_wait_dscnt 0x0" ::: "memory");   // async engine must see LDS writes
    unsigned long long gbase =
        (unsigned long long)(aw_out + ((size_t)bh * 1024 + qb) * 1024);
    unsigned ldsbase = (unsigned)(wid * 16 * PSTR) * 4u;
    for (int r = 0; r < 16; ++r) {
      for (int cc = lane * 4; cc < 1024; cc += 128) {
        unsigned loff = ldsbase + (unsigned)(r * PSTR + cc) * 4u;
        unsigned goff = (unsigned)(r * 1024 + cc) * 4u;
        asm volatile("global_store_async_from_lds_b128 %0, %1, %2"
                     :: "v"(goff), "v"(loff), "s"(gbase) : "memory");
      }
    }
  }

  // --- cv = P @ V (A rebuilt from normalized LDS; padded stride = no conflicts) ---
  const _Float16* vbh = Vt + (size_t)bh * 64 * 1024;
  v8f acc[4] = {};
  for (int ks = 0; ks < 32; ++ks) {
    const float* pr = P + ln * PSTR + ks * 32;
    v16h a;
#pragma unroll
    for (int e = 0; e < 8; ++e) {
      a[e]     = (_Float16)pr[h2 * 8 + e];
      a[8 + e] = (_Float16)pr[16 + h2 * 8 + e];
    }
#pragma unroll
    for (int t = 0; t < 4; ++t) {
      const _Float16* vrow = vbh + (size_t)(t * 16 + ln) * 1024 + ks * 32 + h2 * 8;
      v16h bf = cat16(*(const v8h*)vrow, *(const v8h*)(vrow + 16));
      acc[t] = WMMA_F16(a, bf, acc[t]);
    }
  }

  int hh = bh & 7;
#pragma unroll
  for (int t = 0; t < 4; ++t)
#pragma unroll
    for (int r = 0; r < 8; ++r) {
      int q   = qb + r + 8 * h2;
      int col = hh * 64 + t * 16 + ln;
      cvws[((size_t)b * 1024 + q) * 512 + col] = (_Float16)acc[t][r];
    }
}

// ---------------------------------------------------------------------------
// Launch: d_in = key, klens, value, query, qlens, Wk, Wv, Wq, Wo, bo
// d_out = cv [8*1024*512] f32  ++  aw [8*8*1024*1024] f32
// ---------------------------------------------------------------------------
extern "C" void kernel_launch(void* const* d_in, const int* in_sizes, int n_in,
                              void* d_out, int out_size, void* d_ws, size_t ws_size,
                              hipStream_t stream) {
  const float* key   = (const float*)d_in[0];
  const int*   klens = (const int*)  d_in[1];
  const float* value = (const float*)d_in[2];
  const float* query = (const float*)d_in[3];
  const int*   qlens = (const int*)  d_in[4];
  const float* Wk    = (const float*)d_in[5];
  const float* Wv    = (const float*)d_in[6];
  const float* Wq    = (const float*)d_in[7];
  const float* Wo    = (const float*)d_in[8];
  const float* bo    = (const float*)d_in[9];

  float* out_cv = (float*)d_out;
  float* out_aw = out_cv + (size_t)BATCH * LEN * DMODEL;

  // workspace carve-up
  char* ws = (char*)d_ws;
  const size_t actsz = (size_t)BATCH * LEN * DMODEL * sizeof(_Float16);      // 8 MB
  _Float16* X16  = (_Float16*)(ws);                  // [3][8192*512] q,k,v  (24 MB)
  _Float16* Qh   = (_Float16*)(ws + 3 * actsz);
  _Float16* Kh   = (_Float16*)(ws + 4 * actsz);
  _Float16* Vt   = (_Float16*)(ws + 5 * actsz);
  _Float16* cvws = (_Float16*)(ws + 6 * actsz);
  _Float16* WT   = (_Float16*)(ws + 7 * actsz);      // [4][512][512] q,k,v,o (2 MB)

  // 1) weight transpose/convert + activation f16 conversion
  k_wtrans<<<4096, 256, 0, stream>>>(Wq, Wk, Wv, Wo, WT);
  k_xcvt  <<<12288, 256, 0, stream>>>(query, key, value, X16);

  // 2) projections (16384 tiles -> 4096 blocks x 4 waves)
  k_gemm<<<4096, 128, 0, stream>>>(X16 + 0 * 4194304, WT + 0 * 262144, Qh, nullptr, nullptr, 0);
  k_gemm<<<4096, 128, 0, stream>>>(X16 + 1 * 4194304, WT + 1 * 262144, Kh, nullptr, nullptr, 0);
  k_gemm<<<4096, 128, 0, stream>>>(X16 + 2 * 4194304, WT + 2 * 262144, Vt, nullptr, nullptr, 1);

  // 3) attention core: 4096 strips -> 1024 blocks x 4 waves, 257KB LDS
  size_t lds_bytes = (size_t)4 * 16 * PSTR * sizeof(float);
  k_attn<<<1024, 128, lds_bytes, stream>>>(Qh, Kh, Vt, klens, qlens, out_aw, cvws);

  // 4) output projection (f32 + bias)
  k_gemm<<<4096, 128, 0, stream>>>(cvws, WT + 3 * 262144, nullptr, out_cv, bo, 2);
}
